// LocalFeatureAggregation_1606317769121
// MI455X (gfx1250) — matrix operand order repaired and checked by reading, not compiled, not run on hardware
//
#include <hip/hip_runtime.h>
#include <hip/hip_bf16.h>

typedef _Float16 h16v __attribute__((ext_vector_type(16)));
typedef _Float16 h8v  __attribute__((ext_vector_type(8)));
typedef _Float16 h2v  __attribute__((ext_vector_type(2)));
typedef float    f8v  __attribute__((ext_vector_type(8)));

#define BATCH 4
#define NPTS  8192
#define CH    64
#define KNN   16
#define EPSV  1e-5f
#define WAVES 4
#define LFA_GRID 1024
#define QPB   64          // KNN queries per block
#define SEG   4           // KNN candidate segments per block

// ---------------- CDNA5 helpers ----------------

__device__ __forceinline__ void wave_lds_fence() {
  // Order wave-internal cross-lane LDS traffic explicitly (CDNA5 split counters).
  asm volatile("s_wait_dscnt 0" ::: "memory");
  __builtin_amdgcn_wave_barrier();
}

__device__ __forceinline__ f8v fzero() {
  f8v z;
  #pragma unroll
  for (int i = 0; i < 8; ++i) z[i] = 0.f;
  return z;
}

__device__ __forceinline__ f8v wmma_f16(h16v a, h16v b, f8v c) {
  // D = A(16x32 f16) x B(32x16 f16) + C(16x16 f32)
  return __builtin_amdgcn_wmma_f32_16x16x32_f16(false, a, false, b, (short)0, c,
                                                false, false);
}

// A fragment (16-bit A 16x32 layout): lane<16 row M=lane holds K = {0..7,16..23},
// lane>=16 same row holds K = {8..15,24..31}. Two 16B LDS loads.
__device__ __forceinline__ h16v load_fragA(const _Float16* rowbase, int koff, int hi8) {
  h8v x = *(const h8v*)(rowbase + koff + hi8);
  h8v y = *(const h8v*)(rowbase + koff + 16 + hi8);
  h16v r;
  #pragma unroll
  for (int i = 0; i < 8; ++i) { r[i] = x[i]; r[i + 8] = y[i]; }
  return r;
}

// B fragment (32x16): lanes 0-15 (N=lane) hold K = {0..7,16..23}, lanes 16-31 hold K = {8..15,24..31}.
// Weights stored in LDS as [o][k]; two 16B loads matching the A layout.
__device__ __forceinline__ h16v load_fragB(const _Float16* rowbase, int koff, int k16) {
  h8v x = *(const h8v*)(rowbase + koff + (k16 >> 1));
  h8v y = *(const h8v*)(rowbase + koff + 16 + (k16 >> 1));
  h16v r;
  #pragma unroll
  for (int i = 0; i < 8; ++i) { r[i] = x[i]; r[i + 8] = y[i]; }
  return r;
}

// ---------------- Kernel 1: exact brute-force KNN (top-16 smallest sq-dist) ----
// 256 threads = 64 queries x 4 candidate segments. Each thread scans N/4
// candidates from its own LDS tile buffer, then the 4 partial top-16 lists
// are merged through LDS. Within a wave all lanes share one segment buffer,
// so the inner-loop LDS reads are same-address broadcasts (no bank conflicts).
__global__ __launch_bounds__(256) void knn_topk_kernel(const float* __restrict__ pts,
                                                       int* __restrict__ idx_out) {
  __shared__ float s_cx[SEG][256], s_cy[SEG][256], s_cz[SEG][256], s_cq[SEG][256];
  __shared__ float s_pd[QPB][SEG - 1][16];
  __shared__ int   s_pi[QPB][SEG - 1][16];

  const int b   = blockIdx.y;
  const int tid = threadIdx.x;
  const int q   = tid & (QPB - 1);
  const int s   = tid >> 6;                  // candidate segment 0..3
  const int n   = blockIdx.x * QPB + q;
  const float* pb = pts + (size_t)b * NPTS * 3;

  const float qx = pb[(size_t)n * 3 + 0];
  const float qy = pb[(size_t)n * 3 + 1];
  const float qz = pb[(size_t)n * 3 + 2];
  const float qsq = qx * qx + qy * qy + qz * qz;

  float bd[16]; int bi[16];
  #pragma unroll
  for (int j = 0; j < 16; ++j) { bd[j] = 3.4e38f; bi[j] = 0; }

  const int segbase = s * (NPTS / SEG);
  for (int step = 0; step < (NPTS / SEG) / 256; ++step) {   // 8 steps
    __syncthreads();
    #pragma unroll
    for (int ss = 0; ss < SEG; ++ss) {       // stage 4 tiles of 256 candidates
      const int c = ss * (NPTS / SEG) + step * 256 + tid;
      const float x = pb[(size_t)c * 3 + 0];
      const float y = pb[(size_t)c * 3 + 1];
      const float z = pb[(size_t)c * 3 + 2];
      s_cx[ss][tid] = x; s_cy[ss][tid] = y; s_cz[ss][tid] = z;
      s_cq[ss][tid] = x * x + y * y + z * z;
    }
    __syncthreads();
    const int cbase = segbase + step * 256;
    for (int j = 0; j < 256; ++j) {
      const float d = qsq + s_cq[s][j]
                    - 2.f * (qx * s_cx[s][j] + qy * s_cy[s][j] + qz * s_cz[s][j]);
      if (d < bd[15]) {            // rare after warm-up: guarded insertion sort
        float cd = d; int ci = cbase + j;
        #pragma unroll
        for (int t = 0; t < 16; ++t) {
          if (cd <= bd[t]) {
            float td = bd[t]; bd[t] = cd; cd = td;
            int   ti = bi[t]; bi[t] = ci; ci = ti;
          }
        }
      }
    }
  }

  // merge: segments 1..3 publish their lists, segment 0 folds them in
  if (s > 0) {
    #pragma unroll
    for (int j = 0; j < 16; ++j) {
      s_pd[q][s - 1][j] = bd[j];
      s_pi[q][s - 1][j] = bi[j];
    }
  }
  __syncthreads();
  if (s == 0) {
    const float* pd = &s_pd[q][0][0];
    const int*   pi = &s_pi[q][0][0];
    for (int e = 0; e < (SEG - 1) * 16; ++e) {
      const float d = pd[e];
      if (d < bd[15]) {
        float cd = d; int ci = pi[e];
        #pragma unroll
        for (int t = 0; t < 16; ++t) {
          if (cd <= bd[t]) {
            float td = bd[t]; bd[t] = cd; cd = td;
            int   ti = bi[t]; bi[t] = ci; ci = ti;
          }
        }
      }
    }
    int* o = idx_out + (size_t)(b * NPTS + n) * KNN;
    #pragma unroll
    for (int j = 0; j < 16; ++j) o[j] = bi[j];
  }
}

// ---------------- Kernel 2: gather + 3 WMMA MLPs + max-pool over K -------------
__global__ __launch_bounds__(128) void lfa_wmma_kernel(
    const float* __restrict__ pts,  const float* __restrict__ feat,
    const float* __restrict__ wgeom, const float* __restrict__ g1, const float* __restrict__ b1,
    const float* __restrict__ m1,   const float* __restrict__ v1,
    const float* __restrict__ wsem, const float* __restrict__ g2, const float* __restrict__ b2,
    const float* __restrict__ m2,   const float* __restrict__ v2,
    const float* __restrict__ wfuse, const float* __restrict__ g3, const float* __restrict__ b3,
    const float* __restrict__ m3,   const float* __restrict__ v3,
    const int* __restrict__ knn_idx, float* __restrict__ out, int iters) {

  __shared__ __align__(16) _Float16 s_wsem [64][128];   // [o][k] f16
  __shared__ __align__(16) _Float16 s_wfuse[64][128];   // [o][k] f16
  __shared__ __align__(16) _Float16 s_wgeom[64][32];    // [o][k] f16, k>=6 zero
  __shared__ __align__(16) _Float16 s_rows [WAVES][16][128]; // sem rows, then fused rows
  __shared__ __align__(16) _Float16 s_geom [WAVES][16][32];  // geom rows, zero padded
  __shared__ float s_sc[3][64], s_sh[3][64];            // folded BN scale/shift

  const int tid  = threadIdx.x;
  const int wave = tid >> 5;
  const int lane = tid & 31;
  const int l15  = lane & 15;
  const int hi8  = (lane >> 4) << 3;   // A-frag K offset / C-layout M offset
  const int k16  = (lane >> 4) << 4;   // B-frag K offset

  // ---- stage f16 weights + folded BN params (once per block) ----
  for (int i = tid; i < 64 * 128; i += 128) {
    s_wsem [i >> 7][i & 127] = (_Float16)wsem [i];
    s_wfuse[i >> 7][i & 127] = (_Float16)wfuse[i];
  }
  for (int i = tid; i < 64 * 32; i += 128) {
    const int o = i >> 5, k = i & 31;
    s_wgeom[o][k] = (k < 6) ? (_Float16)wgeom[o * 6 + k] : (_Float16)0.f;
  }
  if (tid < 64) {
    float s = g1[tid] * rsqrtf(v1[tid] + EPSV);
    s_sc[0][tid] = s; s_sh[0][tid] = b1[tid] - m1[tid] * s;
    s = g2[tid] * rsqrtf(v2[tid] + EPSV);
    s_sc[1][tid] = s; s_sh[1][tid] = b2[tid] - m2[tid] * s;
    s = g3[tid] * rsqrtf(v3[tid] + EPSV);
    s_sc[2][tid] = s; s_sh[2][tid] = b3[tid] - m3[tid] * s;
  }
  __syncthreads();

  for (int it = 0; it < iters; ++it) {
    const int p = (it * (int)gridDim.x + (int)blockIdx.x) * WAVES + wave; // one wave = one query point
    const int b = p >> 13;
    const int n = p & (NPTS - 1);
    const float* fb = feat + (size_t)b * NPTS * CH;
    const float* pb = pts  + (size_t)b * NPTS * 3;

    // prefetch next iteration's index row + center-feature row (global_prefetch_b8)
    if (it + 1 < iters) {
      const int pn = ((it + 1) * (int)gridDim.x + (int)blockIdx.x) * WAVES + wave;
      const int bn_ = pn >> 13, nn_ = pn & (NPTS - 1);
      __builtin_prefetch(knn_idx + (size_t)(bn_ * NPTS + nn_) * KNN, 0, 1);
      __builtin_prefetch(feat + ((size_t)bn_ * NPTS + nn_) * CH, 0, 1);
    }

    const int myNbr = knn_idx[(size_t)(b * NPTS + n) * KNN + l15]; // lanes 0-15 valid

    // center features: two channels per lane
    const float2 cc = *(const float2*)(fb + (size_t)n * CH + lane * 2);

    // ---- sem A rows: [ctr(64) | nbr-ctr(64)] ----
    #pragma unroll 4
    for (int m = 0; m < 16; ++m) {
      const int nb = __shfl(myNbr, m);
      const float2 nf = *(const float2*)(fb + (size_t)nb * CH + lane * 2);
      *(h2v*)&s_rows[wave][m][lane * 2]      = (h2v){(_Float16)cc.x, (_Float16)cc.y};
      *(h2v*)&s_rows[wave][m][64 + lane * 2] = (h2v){(_Float16)(nf.x - cc.x),
                                                     (_Float16)(nf.y - cc.y)};
    }

    // ---- geom A rows: [ctr.xyz | nbr-ctr | zeros to K=32] ----
    for (int t = lane; t < 16 * 32; t += 32)
      s_geom[wave][t >> 5][t & 31] = (_Float16)0.f;
    {
      const float px = pb[(size_t)n * 3 + 0];
      const float py = pb[(size_t)n * 3 + 1];
      const float pz = pb[(size_t)n * 3 + 2];
      if (lane < 16) {
        const float nx = pb[(size_t)myNbr * 3 + 0];
        const float ny = pb[(size_t)myNbr * 3 + 1];
        const float nz = pb[(size_t)myNbr * 3 + 2];
        s_geom[wave][lane][0] = (_Float16)px;
        s_geom[wave][lane][1] = (_Float16)py;
        s_geom[wave][lane][2] = (_Float16)pz;
        s_geom[wave][lane][3] = (_Float16)(nx - px);
        s_geom[wave][lane][4] = (_Float16)(ny - py);
        s_geom[wave][lane][5] = (_Float16)(nz - pz);
      }
    }
    wave_lds_fence();

    // ---- geom MLP: 16x32 @ 32x64 (4 WMMAs) ----
    f8v accg[4];
    {
      const h16v a = load_fragA(&s_geom[wave][l15][0], 0, hi8);
      #pragma unroll
      for (int nt = 0; nt < 4; ++nt) {
        const h16v bf = load_fragB(&s_wgeom[nt * 16 + l15][0], 0, k16);
        accg[nt] = wmma_f16(a, bf, fzero());
      }
    }

    // ---- sem MLP: 16x128 @ 128x64 (16 WMMAs) ----
    f8v accs[4];
    #pragma unroll
    for (int nt = 0; nt < 4; ++nt) accs[nt] = fzero();
    #pragma unroll
    for (int kc = 0; kc < 4; ++kc) {
      const h16v a = load_fragA(&s_rows[wave][l15][0], kc * 32, hi8);
      #pragma unroll
      for (int nt = 0; nt < 4; ++nt) {
        const h16v bf = load_fragB(&s_wsem[nt * 16 + l15][0], kc * 32, k16);
        accs[nt] = wmma_f16(a, bf, accs[nt]);
      }
    }

    // ---- BN+ReLU, write fused rows [geom_feat | sem_feat] (C-layout -> A rows) ----
    #pragma unroll
    for (int nt = 0; nt < 4; ++nt) {
      const int o = nt * 16 + l15;
      const float sc1 = s_sc[0][o], sh1 = s_sh[0][o];
      const float sc2 = s_sc[1][o], sh2 = s_sh[1][o];
      #pragma unroll
      for (int r = 0; r < 8; ++r) {
        float yg = fmaf(accg[nt][r], sc1, sh1); yg = yg > 0.f ? yg : 0.f;
        float ys = fmaf(accs[nt][r], sc2, sh2); ys = ys > 0.f ? ys : 0.f;
        s_rows[wave][r + hi8][o]      = (_Float16)yg;
        s_rows[wave][r + hi8][64 + o] = (_Float16)ys;
      }
    }
    wave_lds_fence();

    // ---- fuse MLP: 16x128 @ 128x64 (16 WMMAs) ----
    f8v accf[4];
    #pragma unroll
    for (int nt = 0; nt < 4; ++nt) accf[nt] = fzero();
    #pragma unroll
    for (int kc = 0; kc < 4; ++kc) {
      const h16v a = load_fragA(&s_rows[wave][l15][0], kc * 32, hi8);
      #pragma unroll
      for (int nt = 0; nt < 4; ++nt) {
        const h16v bf = load_fragB(&s_wfuse[nt * 16 + l15][0], kc * 32, k16);
        accf[nt] = wmma_f16(a, bf, accf[nt]);
      }
    }

    // ---- BN+ReLU + max over the 16 neighbor rows, store out[b,n,0:64] ----
    #pragma unroll
    for (int nt = 0; nt < 4; ++nt) {
      const int o = nt * 16 + l15;
      const float sc3 = s_sc[2][o], sh3 = s_sh[2][o];
      float mx = -3.4e38f;
      #pragma unroll
      for (int r = 0; r < 8; ++r) {
        float y = fmaf(accf[nt][r], sc3, sh3); y = y > 0.f ? y : 0.f;
        mx = y > mx ? y : mx;
      }
      const float ox = __shfl_xor(mx, 16);   // combine M=0..7 with M=8..15 halves
      mx = ox > mx ? ox : mx;
      if (lane < 16)
        out[(size_t)(b * NPTS + n) * CH + o] = mx;
    }
    wave_lds_fence();   // protect staging before next iteration overwrites it
  }
}

extern "C" void kernel_launch(void* const* d_in, const int* in_sizes, int n_in,
                              void* d_out, int out_size, void* d_ws, size_t ws_size,
                              hipStream_t stream) {
  const float* pts    = (const float*)d_in[0];
  const float* feat   = (const float*)d_in[1];
  const float* wgeom  = (const float*)d_in[2];
  const float* g1     = (const float*)d_in[3];
  const float* b1     = (const float*)d_in[4];
  const float* m1     = (const float*)d_in[5];
  const float* v1     = (const float*)d_in[6];
  const float* wsem   = (const float*)d_in[7];
  const float* g2     = (const float*)d_in[8];
  const float* b2     = (const float*)d_in[9];
  const float* m2     = (const float*)d_in[10];
  const float* v2     = (const float*)d_in[11];
  const float* wfuse  = (const float*)d_in[12];
  const float* g3     = (const float*)d_in[13];
  const float* b3     = (const float*)d_in[14];
  const float* m3     = (const float*)d_in[15];
  const float* v3     = (const float*)d_in[16];
  float* out = (float*)d_out;
  int* knn_idx = (int*)d_ws;   // B*N*K ints = 2 MB scratch

  dim3 kgrid(NPTS / QPB, BATCH);             // 128 x 4 = 512 blocks
  knn_topk_kernel<<<kgrid, 256, 0, stream>>>(pts, knn_idx);

  const int iters = (BATCH * NPTS) / (LFA_GRID * WAVES);  // 8, uniform per block
  lfa_wmma_kernel<<<LFA_GRID, 128, 0, stream>>>(
      pts, feat, wgeom, g1, b1, m1, v1, wsem, g2, b2, m2, v2,
      wfuse, g3, b3, m3, v3, knn_idx, out, iters);
}
